// PointNet2SSGEncoder_28209345200318
// MI455X (gfx1250) — matrix operand order repaired and checked by reading, not compiled, not run on hardware
//
#include <hip/hip_runtime.h>
#include <hip/hip_bf16.h>

// ---------------------------------------------------------------------------
// Types for gfx1250 WMMA (wave32).
// ---------------------------------------------------------------------------
typedef __attribute__((ext_vector_type(16))) __bf16 v16bf;
typedef __attribute__((ext_vector_type(8)))  float  v8f;

struct __align__(16) U4 { unsigned int x, y, z, w; };
union FragBF { v16bf v; U4 u4[2]; };

__device__ __forceinline__ unsigned short f2bf(float f) {
    // round-to-nearest-even fp32 -> bf16 (bit pattern)
    unsigned int u = __float_as_uint(f);
    u += 0x7fffu + ((u >> 16) & 1u);
    return (unsigned short)(u >> 16);
}

// ---------------------------------------------------------------------------
// Deinterleave pointcloud (B*N, 6) -> xyz (B*N,3), feats (B*N,3)
// ---------------------------------------------------------------------------
__global__ void split_pc_kernel(const float* __restrict__ pc,
                                float* __restrict__ xyz,
                                float* __restrict__ f, int total) {
    int i = blockIdx.x * 256 + threadIdx.x;
    if (i < total) {
        xyz[i * 3 + 0] = pc[i * 6 + 0];
        xyz[i * 3 + 1] = pc[i * 6 + 1];
        xyz[i * 3 + 2] = pc[i * 6 + 2];
        f[i * 3 + 0] = pc[i * 6 + 3];
        f[i * 3 + 1] = pc[i * 6 + 4];
        f[i * 3 + 2] = pc[i * 6 + 5];
    }
}

// ---------------------------------------------------------------------------
// Farthest point sampling: one block per batch, whole point set LDS-resident.
// Dynamic LDS: 4*N floats (x,y,z,dist) -> 128KB for N=8192 (fits 320KB WGP).
// ---------------------------------------------------------------------------
__global__ __launch_bounds__(256) void fps_kernel(const float* __restrict__ xyz,
                                                  int N, int npoint,
                                                  int* __restrict__ fidx,
                                                  float* __restrict__ new_xyz) {
    extern __shared__ float sm[];
    float* sx = sm;
    float* sy = sm + N;
    float* sz = sm + 2 * N;
    float* sd = sm + 3 * N;
    __shared__ float rv[256];
    __shared__ int   ri[256];
    __shared__ int   curFar;

    int b = blockIdx.x;
    int tid = threadIdx.x;
    const float* base = xyz + (size_t)b * N * 3;
    for (int j = tid; j < N; j += 256) {
        sx[j] = base[j * 3 + 0];
        sy[j] = base[j * 3 + 1];
        sz[j] = base[j * 3 + 2];
        sd[j] = 1e10f;
    }
    if (tid == 0) curFar = 0;
    __syncthreads();

    for (int i = 0; i < npoint; ++i) {
        int far = curFar;
        float cx = sx[far], cy = sy[far], cz = sz[far];
        if (tid == 0) {
            fidx[b * npoint + i] = far;
            new_xyz[((size_t)b * npoint + i) * 3 + 0] = cx;
            new_xyz[((size_t)b * npoint + i) * 3 + 1] = cy;
            new_xyz[((size_t)b * npoint + i) * 3 + 2] = cz;
        }
        float bestv = -1.0f;
        int   besti = 0;
        for (int j = tid; j < N; j += 256) {
            float dx = sx[j] - cx, dy = sy[j] - cy, dz = sz[j] - cz;
            float d = dx * dx + dy * dy + dz * dz;
            float dd = sd[j];
            dd = d < dd ? d : dd;
            sd[j] = dd;
            if (dd > bestv) { bestv = dd; besti = j; }
        }
        rv[tid] = bestv; ri[tid] = besti;
        __syncthreads();
        for (int s = 128; s > 0; s >>= 1) {
            if (tid < s) {
                if (rv[tid + s] > rv[tid] ||
                    (rv[tid + s] == rv[tid] && ri[tid + s] < ri[tid])) {
                    rv[tid] = rv[tid + s]; ri[tid] = ri[tid + s];
                }
            }
            __syncthreads();
        }
        if (tid == 0) curFar = ri[0];
        __syncthreads();
    }
}

// ---------------------------------------------------------------------------
// Ball query + gather + bf16 pack. One wave32 per query; ordered index
// extraction via ballot (matches reference's sorted-index-first-64 semantics).
// ---------------------------------------------------------------------------
__global__ __launch_bounds__(256) void group_kernel(
    const float* __restrict__ xyz, const float* __restrict__ feats,
    const float* __restrict__ new_xyz, unsigned short* __restrict__ A0,
    int B, int N, int S, int nsample, int Cf, int Kpad, float r2) {
    __shared__ int idxbuf[8][64];
    int wave = threadIdx.x >> 5;
    int lane = threadIdx.x & 31;
    int q = blockIdx.x * 8 + wave;   // B*S is a multiple of 8 for all calls
    int b = q / S;

    float cx = new_xyz[(size_t)q * 3 + 0];
    float cy = new_xyz[(size_t)q * 3 + 1];
    float cz = new_xyz[(size_t)q * 3 + 2];
    const float* xb = xyz + (size_t)b * N * 3;

    int count = 0;
    for (int j0 = 0; j0 < N && count < nsample; j0 += 32) {
        int j = j0 + lane;
        bool hit = false;
        if (j < N) {
            float dx = xb[j * 3 + 0] - cx;
            float dy = xb[j * 3 + 1] - cy;
            float dz = xb[j * 3 + 2] - cz;
            hit = (dx * dx + dy * dy + dz * dz) < r2;
        }
        unsigned m = (unsigned)__ballot(hit);
        if (hit) {
            int pos = count + __popc(m & ((1u << lane) - 1u));
            if (pos < nsample) idxbuf[wave][pos] = j;
        }
        count += __popc(m);
    }
    if (count > nsample) count = nsample;
    __syncthreads();
    int first = idxbuf[wave][0];   // query point itself always hits (d2=0)
    for (int t = count + lane; t < nsample; t += 32) idxbuf[wave][t] = first;
    __syncthreads();

    for (int n = lane; n < nsample; n += 32) {
        int idx = idxbuf[wave][n];
        size_t row = (size_t)q * nsample + n;
        unsigned short* dst = A0 + row * Kpad;
        dst[0] = f2bf(xb[idx * 3 + 0] - cx);
        dst[1] = f2bf(xb[idx * 3 + 1] - cy);
        dst[2] = f2bf(xb[idx * 3 + 2] - cz);
        const float* fp = feats + ((size_t)b * N + idx) * Cf;
        for (int c = 0; c < Cf; ++c) dst[3 + c] = f2bf(fp[c]);
        for (int c = 3 + Cf; c < Kpad; ++c) dst[c] = 0;
    }
}

// ---------------------------------------------------------------------------
// Stage-3 input build: concat(xyz(3), feats(Cf)) -> bf16, zero pad to Kpad.
// ---------------------------------------------------------------------------
__global__ void concat3_kernel(const float* __restrict__ xyz,
                               const float* __restrict__ feats,
                               unsigned short* __restrict__ A0,
                               int rows, int Cf, int Kpad) {
    int i = blockIdx.x * 256 + threadIdx.x;
    int row = i / Kpad, c = i % Kpad;
    if (row < rows) {
        float v;
        if (c < 3) v = xyz[row * 3 + c];
        else if (c < 3 + Cf) v = feats[(size_t)row * Cf + (c - 3)];
        else v = 0.0f;
        A0[(size_t)row * Kpad + c] = f2bf(v);
    }
}

// ---------------------------------------------------------------------------
// Pre-pack weights (Cout, Cin) fp32 -> (Cout, Kpad) bf16 (zero K-padding).
// Removes all masking/conversion from the GEMM hot loop.
// ---------------------------------------------------------------------------
__global__ void pack_w_kernel(const float* __restrict__ W,
                              unsigned short* __restrict__ Wb,
                              int Cout, int Cin, int Kpad) {
    int i = blockIdx.x * 256 + threadIdx.x;
    if (i < Cout * Kpad) {
        int row = i / Kpad, c = i % Kpad;
        Wb[i] = (c < Cin) ? f2bf(W[(size_t)row * Cin + c]) : (unsigned short)0;
    }
}

// ---------------------------------------------------------------------------
// bf16 WMMA GEMM: Y[M,Cout] = A[M,K] * Wb[Cout,K]^T  (fp32 accumulate).
//  - Block: 256 thr = 8 waves; block tile 128x64; each wave owns a 32x32
//    output subtile -> 4x v_wmma_f32_16x16x32_bf16 per K-step.
//  - Double-buffered LDS tiles; A/B tiles delivered by
//    GLOBAL_LOAD_ASYNC_TO_LDS_B128 (ASYNCcnt) when bf16-prepacked, or by a
//    float4-vectorized fused batchnorm+ReLU+bf16 convert for the A tile of
//    interior layers (Cin always a multiple of 4 there -> aligned b128 loads).
//  - One barrier per K-step; next tile's copies issued before compute.
// ---------------------------------------------------------------------------
__global__ __launch_bounds__(256) void gemm_bn_kernel(
    const unsigned short* __restrict__ A0,  // bf16 bits or nullptr
    const float* __restrict__ Yprev,        // fp32 (used when A0==nullptr)
    const float* __restrict__ mu, const float* __restrict__ rs,
    const float* __restrict__ gamma, const float* __restrict__ beta,
    const unsigned short* __restrict__ Wb,  // (Cout, Kpad) bf16
    float* __restrict__ Y,                  // (M, Cout) fp32
    int M, int Cin, int Kpad, int Cout) {
    __shared__ __align__(16) unsigned short As[2][128][32];  // 16 KB
    __shared__ __align__(16) unsigned short Bs[2][64][32];   //  8 KB

    int tid = threadIdx.x;
    int wave = tid >> 5, lane = tid & 31;
    int rowBase = blockIdx.x * 128;
    int colBase = blockIdx.y * 64;
    int m0 = (wave >> 1) * 32;   // 4 waves along M
    int n0 = (wave & 1) * 32;    // 2 waves along N

    v8f acc00 = {}, acc01 = {}, acc10 = {}, acc11 = {};

    int aRow = tid >> 1;           // 0..127, 16 bf16 per thread
    int aK   = (tid & 1) * 16;
    int bRow = tid >> 2;           // 0..63, 8 bf16 per thread
    int bK   = (tid & 3) * 8;
    int numK = Kpad >> 5;

    auto load_tile = [&](int kt, int buf) {
        int kBase = kt * 32;
        // ---- A tile 128x32 ----
        if (A0) {
            const unsigned short* src =
                A0 + (size_t)(rowBase + aRow) * Kpad + kBase + aK;
            unsigned lds0 = (unsigned)(size_t)&As[buf][aRow][aK];
            unsigned long long ga = (unsigned long long)(size_t)src;
            asm volatile(
                "global_load_async_to_lds_b128 %0, %1, off\n\t"
                "global_load_async_to_lds_b128 %2, %3, off"
                :: "v"(lds0), "v"(ga), "v"(lds0 + 16u), "v"(ga + 16ull)
                : "memory");
        } else {
            const float* src =
                Yprev + (size_t)(rowBase + aRow) * Cin + kBase + aK;
            if (kBase + 32 < Kpad) __builtin_prefetch(src + 32, 0, 1);
            const float4* v4 = (const float4*)src;   // Cin % 4 == 0, aligned
            float4 p0 = v4[0], p1 = v4[1], p2 = v4[2], p3 = v4[3];
            float xv[16];
            xv[0] = p0.x;  xv[1] = p0.y;  xv[2] = p0.z;  xv[3] = p0.w;
            xv[4] = p1.x;  xv[5] = p1.y;  xv[6] = p1.z;  xv[7] = p1.w;
            xv[8] = p2.x;  xv[9] = p2.y;  xv[10] = p2.z; xv[11] = p2.w;
            xv[12] = p3.x; xv[13] = p3.y; xv[14] = p3.z; xv[15] = p3.w;
            __align__(16) unsigned short tmp[16];
#pragma unroll
            for (int j = 0; j < 16; ++j) {
                int c = kBase + aK + j;
                float x = (xv[j] - mu[c]) * rs[c] * gamma[c] + beta[c];
                x = x > 0.0f ? x : 0.0f;
                tmp[j] = f2bf(x);
            }
            *(U4*)&As[buf][aRow][aK]     = *(const U4*)&tmp[0];
            *(U4*)&As[buf][aRow][aK + 8] = *(const U4*)&tmp[8];
        }
        // ---- B tile 64x32 (prepacked bf16 -> pure async copy) ----
        {
            const unsigned short* wsrc =
                Wb + (size_t)(colBase + bRow) * Kpad + kBase + bK;
            unsigned ldsb = (unsigned)(size_t)&Bs[buf][bRow][bK];
            asm volatile(
                "global_load_async_to_lds_b128 %0, %1, off"
                :: "v"(ldsb), "v"((unsigned long long)(size_t)wsrc)
                : "memory");
        }
    };

    load_tile(0, 0);
    int cur = 0;
    for (int kt = 0; kt < numK; ++kt) {
        asm volatile("s_wait_asynccnt 0x0" ::: "memory");
        __syncthreads();                       // tile[cur] ready block-wide
        if (kt + 1 < numK) load_tile(kt + 1, cur ^ 1);

        // ---- fragments (A 16x32: lane&15=M, lane>=16 -> K halves {8..,24..};
        //       B 32x16: lane&15=N, same K halves as A) ----
        int ka = (lane >> 4) * 8;
        int kb = (lane >> 4) * 8;
        int mA = m0 + (lane & 15);
        int nB = n0 + (lane & 15);
        FragBF fa0, fa1, fb0, fb1;
        fa0.u4[0] = *(const U4*)&As[cur][mA][ka];
        fa0.u4[1] = *(const U4*)&As[cur][mA][ka + 16];
        fa1.u4[0] = *(const U4*)&As[cur][mA + 16][ka];
        fa1.u4[1] = *(const U4*)&As[cur][mA + 16][ka + 16];
        fb0.u4[0] = *(const U4*)&Bs[cur][nB][kb];
        fb0.u4[1] = *(const U4*)&Bs[cur][nB][kb + 16];
        fb1.u4[0] = *(const U4*)&Bs[cur][nB + 16][kb];
        fb1.u4[1] = *(const U4*)&Bs[cur][nB + 16][kb + 16];

        acc00 = __builtin_amdgcn_wmma_f32_16x16x32_bf16(
            false, fa0.v, false, fb0.v, (short)0, acc00, false, false);
        acc01 = __builtin_amdgcn_wmma_f32_16x16x32_bf16(
            false, fa0.v, false, fb1.v, (short)0, acc01, false, false);
        acc10 = __builtin_amdgcn_wmma_f32_16x16x32_bf16(
            false, fa1.v, false, fb0.v, (short)0, acc10, false, false);
        acc11 = __builtin_amdgcn_wmma_f32_16x16x32_bf16(
            false, fa1.v, false, fb1.v, (short)0, acc11, false, false);
        cur ^= 1;
    }

    // ---- store D: VGPR i holds M=i (lanes 0-15) / M=8+i (lanes 16-31) ----
    int col = colBase + n0 + (lane & 15);
    int rHalf = (lane >> 4) * 8;
#pragma unroll
    for (int i = 0; i < 8; ++i) {
        int r0 = rowBase + m0 + rHalf + i;
        Y[(size_t)r0 * Cout + col]              = acc00[i];
        Y[(size_t)r0 * Cout + col + 16]         = acc01[i];
        Y[(size_t)(r0 + 16) * Cout + col]       = acc10[i];
        Y[(size_t)(r0 + 16) * Cout + col + 16]  = acc11[i];
    }
}

// ---------------------------------------------------------------------------
// Per-channel sum / sumsq over all M rows. sums[0..C)=sum, sums[C..2C)=sumsq.
// ---------------------------------------------------------------------------
__global__ void zero_kernel(float* __restrict__ p, int n) {
    int i = blockIdx.x * 256 + threadIdx.x;
    if (i < n) p[i] = 0.0f;
}

__global__ __launch_bounds__(256) void stats_kernel(const float* __restrict__ Y,
                                                    float* __restrict__ sums,
                                                    int M, int C) {
    int c = blockIdx.x;
    int tid = threadIdx.x;
    __shared__ float s1[256], s2[256];
    float a = 0.0f, b = 0.0f;
    for (int r = blockIdx.y * 256 + tid; r < M; r += gridDim.y * 256) {
        float v = Y[(size_t)r * C + c];
        a += v; b += v * v;
    }
    s1[tid] = a; s2[tid] = b;
    __syncthreads();
    for (int s = 128; s > 0; s >>= 1) {
        if (tid < s) { s1[tid] += s1[tid + s]; s2[tid] += s2[tid + s]; }
        __syncthreads();
    }
    if (tid == 0) {
        atomicAdd(&sums[c], s1[0]);
        atomicAdd(&sums[C + c], s2[0]);
    }
}

__global__ void finalize_stats_kernel(const float* __restrict__ sums,
                                      float* __restrict__ mu,
                                      float* __restrict__ rs,
                                      int C, float invM) {
    int c = blockIdx.x * 256 + threadIdx.x;
    if (c < C) {
        float m = sums[c] * invM;
        float v = sums[C + c] * invM - m * m;
        mu[c] = m;
        rs[c] = rsqrtf(v + 1e-5f);
    }
}

// ---------------------------------------------------------------------------
// Final per-stage epilogue: batchnorm + ReLU + max over nsample neighbors.
// ---------------------------------------------------------------------------
__global__ void norm_relu_max_kernel(const float* __restrict__ Y,
                                     const float* __restrict__ mu,
                                     const float* __restrict__ rs,
                                     const float* __restrict__ g,
                                     const float* __restrict__ b,
                                     float* __restrict__ out,
                                     int rowsOut, int nsample, int C) {
    int c = blockIdx.y * 256 + threadIdx.x;
    int r = blockIdx.x;
    if (c >= C) return;
    float muc = mu[c], rsc = rs[c], gc = g[c], bc = b[c];
    const float* base = Y + (size_t)r * nsample * C + c;
    float best = -1e30f;
    for (int n = 0; n < nsample; ++n) {
        float x = base[(size_t)n * C];
        x = (x - muc) * rsc * gc + bc;
        x = x > 0.0f ? x : 0.0f;
        best = fmaxf(best, x);
    }
    out[(size_t)r * C + c] = best;
}

// ---------------------------------------------------------------------------
// Host-side layer runner (pack W -> GEMM -> stats -> finalize), on `stream`.
// ---------------------------------------------------------------------------
static void run_layer(const unsigned short* A0, const float* Yprev,
                      const float* muPrev, const float* rsPrev,
                      const float* gPrev, const float* bPrev,
                      const float* W, unsigned short* Wb, float* Y,
                      float* sums, float* mu, float* rs,
                      int M, int Cin, int Kpad, int Cout, hipStream_t stream) {
    pack_w_kernel<<<(Cout * Kpad + 255) / 256, 256, 0, stream>>>(
        W, Wb, Cout, Cin, Kpad);
    dim3 g(M / 128, Cout / 64);
    gemm_bn_kernel<<<g, 256, 0, stream>>>(A0, Yprev, muPrev, rsPrev, gPrev,
                                          bPrev, Wb, Y, M, Cin, Kpad, Cout);
    zero_kernel<<<(2 * Cout + 255) / 256, 256, 0, stream>>>(sums, 2 * Cout);
    int chunks = (M + 256 * 32 - 1) / (256 * 32);
    if (chunks < 1) chunks = 1;
    if (chunks > 64) chunks = 64;
    stats_kernel<<<dim3(Cout, chunks), 256, 0, stream>>>(Y, sums, M, Cout);
    finalize_stats_kernel<<<(Cout + 255) / 256, 256, 0, stream>>>(
        sums, mu, rs, Cout, 1.0f / (float)M);
}

// ---------------------------------------------------------------------------
// kernel_launch
// Input order: [0]=pointcloud(16,8192,6);
//   stage1: [1]W(64,6) [2]g [3]b  [4]W(64,64) [5]g [6]b  [7]W(128,64) [8]g [9]b
//   stage2: [10]W(128,131)... [13]W(128,128)... [16]W(256,128)...
//   stage3: [19]W(256,259)... [22]W(512,256)... [25]W(1024,512) [26]g [27]b
// Output: (16, 1024) fp32.
// ---------------------------------------------------------------------------
extern "C" void kernel_launch(void* const* d_in, const int* in_sizes, int n_in,
                              void* d_out, int out_size, void* d_ws,
                              size_t ws_size, hipStream_t stream) {
    (void)in_sizes; (void)n_in; (void)out_size; (void)ws_size;
    const int B = 16, N = 8192;
    const int S1 = 512, S2 = 128, NS = 64;
    const int M1 = B * S1 * NS;      // 524288
    const int M2 = B * S2 * NS;      // 131072
    const int M3 = B * S2;           // 2048

    char* ws = (char*)d_ws;
    size_t off = 0;
    auto alloc = [&](size_t bytes) -> void* {
        off = (off + 255) & ~(size_t)255;
        void* p = ws + off;
        off += bytes;
        return p;
    };

    float* xyz0   = (float*)alloc((size_t)B * N * 3 * 4);
    float* f0     = (float*)alloc((size_t)B * N * 3 * 4);
    int*   fidx1  = (int*)alloc((size_t)B * S1 * 4);
    float* nxyz1  = (float*)alloc((size_t)B * S1 * 3 * 4);
    int*   fidx2  = (int*)alloc((size_t)B * S2 * 4);
    float* nxyz2  = (float*)alloc((size_t)B * S2 * 3 * 4);
    unsigned short* A1 = (unsigned short*)alloc((size_t)M1 * 32 * 2);
    unsigned short* A2 = (unsigned short*)alloc((size_t)M2 * 160 * 2);
    unsigned short* A3 = (unsigned short*)alloc((size_t)M3 * 288 * 2);
    unsigned short* Wb = (unsigned short*)alloc((size_t)1024 * 512 * 2);
    float* feats1 = (float*)alloc((size_t)B * S1 * 128 * 4);
    float* feats2 = (float*)alloc((size_t)B * S2 * 256 * 4);
    float* sums   = (float*)alloc((size_t)2 * 1024 * 4);
    float* mu     = (float*)alloc((size_t)1024 * 4);
    float* rs     = (float*)alloc((size_t)1024 * 4);
    float* Ya     = (float*)alloc((size_t)M1 * 128 * 4);  // 268 MB
    float* Yb     = (float*)alloc((size_t)M1 * 64 * 4);   // 134 MB

    const float* pc = (const float*)d_in[0];
    auto F = [&](int i) { return (const float*)d_in[i]; };

    // ---- Stage 1 ----
    split_pc_kernel<<<(B * N + 255) / 256, 256, 0, stream>>>(pc, xyz0, f0, B * N);
    fps_kernel<<<B, 256, 4 * N * sizeof(float), stream>>>(xyz0, N, S1, fidx1, nxyz1);
    group_kernel<<<(B * S1) / 8, 256, 0, stream>>>(xyz0, f0, nxyz1, A1,
                                                   B, N, S1, NS, 3, 32, 0.04f);
    run_layer(A1, nullptr, nullptr, nullptr, nullptr, nullptr,
              F(1), Wb, Ya, sums, mu, rs, M1, 6, 32, 64, stream);
    run_layer(nullptr, Ya, mu, rs, F(2), F(3),
              F(4), Wb, Yb, sums, mu, rs, M1, 64, 64, 64, stream);
    run_layer(nullptr, Yb, mu, rs, F(5), F(6),
              F(7), Wb, Ya, sums, mu, rs, M1, 64, 64, 128, stream);
    norm_relu_max_kernel<<<dim3(B * S1, 1), 256, 0, stream>>>(
        Ya, mu, rs, F(8), F(9), feats1, B * S1, NS, 128);

    // ---- Stage 2 ----
    fps_kernel<<<B, 256, 4 * S1 * sizeof(float), stream>>>(nxyz1, S1, S2, fidx2, nxyz2);
    group_kernel<<<(B * S2) / 8, 256, 0, stream>>>(nxyz1, feats1, nxyz2, A2,
                                                   B, S1, S2, NS, 128, 160, 0.16f);
    run_layer(A2, nullptr, nullptr, nullptr, nullptr, nullptr,
              F(10), Wb, Ya, sums, mu, rs, M2, 131, 160, 128, stream);
    run_layer(nullptr, Ya, mu, rs, F(11), F(12),
              F(13), Wb, Yb, sums, mu, rs, M2, 128, 128, 128, stream);
    run_layer(nullptr, Yb, mu, rs, F(14), F(15),
              F(16), Wb, Ya, sums, mu, rs, M2, 128, 128, 256, stream);
    norm_relu_max_kernel<<<dim3(B * S2, 1), 256, 0, stream>>>(
        Ya, mu, rs, F(17), F(18), feats2, B * S2, NS, 256);

    // ---- Stage 3 (global MLP over 128 points per batch) ----
    concat3_kernel<<<((size_t)M3 * 288 + 255) / 256, 256, 0, stream>>>(
        nxyz2, feats2, A3, M3, 256, 288);
    run_layer(A3, nullptr, nullptr, nullptr, nullptr, nullptr,
              F(19), Wb, Ya, sums, mu, rs, M3, 259, 288, 256, stream);
    run_layer(nullptr, Ya, mu, rs, F(20), F(21),
              F(22), Wb, Yb, sums, mu, rs, M3, 256, 256, 512, stream);
    run_layer(nullptr, Yb, mu, rs, F(23), F(24),
              F(25), Wb, Ya, sums, mu, rs, M3, 512, 512, 1024, stream);
    norm_relu_max_kernel<<<dim3(B, 4), 256, 0, stream>>>(
        Ya, mu, rs, F(26), F(27), (float*)d_out, B, S2, 1024);
}
